// EmotionRnn_37804302139696
// MI455X (gfx1250) — matrix operand order repaired and checked by reading, not compiled, not run on hardware
//
#include <hip/hip_runtime.h>

// ---------------------------------------------------------------------------
// Problem constants (from reference)
// ---------------------------------------------------------------------------
static constexpr int T_STEPS = 50;
static constexpr int BATCH   = 4096;
static constexpr int IN_DIM  = 256;
static constexpr int HID     = 512;
static constexpr int GDIM    = 4 * HID;     // 2048 gate columns
static constexpr int OUT_DIM = 2;

// Tiling
static constexpr int BT      = 32;          // batch rows per workgroup (2 WMMA row tiles)
static constexpr int XS_STR  = IN_DIM + 8;  // LDS f16 row stride (pad -> conflict-free b128)
static constexpr int H_STR   = HID + 8;     // LDS f16 row stride for h
static constexpr int C_STR   = HID + 4;     // LDS f32 row stride for c

// LDS layout (bytes)
static constexpr int XS_BYTES   = BT * XS_STR * 2;           // 16896
static constexpr int HBUF_BYTES = BT * H_STR * 2;            // 33280 (x4: h1[2], h2[2])
static constexpr int CBUF_BYTES = BT * C_STR * 4;            // 66048 (x2: c1, c2)
static constexpr int OFF_H      = XS_BYTES;
static constexpr int OFF_C      = OFF_H + 4 * HBUF_BYTES;    // 150016
static constexpr int OFF_LRED   = OFF_C + 2 * CBUF_BYTES;    // 282112
static constexpr int SMEM_BYTES = OFF_LRED + 256 * 2 * 4;    // 284160 (< 320KB WGP LDS)

// Packed f16 weight offsets inside d_ws (in _Float16 units)
static constexpr long W1P_OFF = 0;
static constexpr long U1P_OFF = (long)IN_DIM * GDIM;                 // 524288
static constexpr long W2P_OFF = U1P_OFF + (long)HID * GDIM;         // 1572864
static constexpr long U2P_OFF = W2P_OFF + (long)HID * GDIM;         // 2621440
static constexpr long PACK_F16_TOTAL = U2P_OFF + (long)HID * GDIM;  // 3670016 (~7 MB)

typedef __attribute__((ext_vector_type(16))) _Float16 v16h;
typedef __attribute__((ext_vector_type(8)))  float    v8f;
typedef __attribute__((ext_vector_type(4)))  int      i32x4;

union f16frag { i32x4 i[2]; v16h v; };

// ---------------------------------------------------------------------------
// Weight prepack: fp32 row-major [K x 2048] -> f16 WMMA-B fragment tiles.
// Tile (nt, kt) = B block [kt*32 .. +32) x [nt*16 .. +16), stored as 1KB:
// dword (lane*8 + reg) holds f16 pair (K = k0, k0+1) for column N0 + (lane&15)
// with k0 = kt*32 + (reg>>2)*16 + (lane>>4)*8 + 2*(reg&3)   (B-matrix lane/K striping).
// Tiles ordered nt-major so the k-loop streams contiguous 1KB chunks.
// ---------------------------------------------------------------------------
__global__ void __launch_bounds__(256) prepack_kernel(
    const float* __restrict__ W1, const float* __restrict__ U1,
    const float* __restrict__ W2, const float* __restrict__ U2,
    _Float16* __restrict__ dst)
{
  long d = (long)blockIdx.x * 256 + threadIdx.x;   // output dword id
  const long dW1 = (long)IN_DIM * GDIM / 2;        // dwords per matrix = K*1024
  const long dU  = (long)HID * GDIM / 2;
  const float* src; int K; long base;
  if      (d <  dW1)          { src = W1; K = IN_DIM; base = 0; }
  else if (d <  dW1 + dU)     { src = U1; K = HID;    base = dW1; }
  else if (d <  dW1 + 2 * dU) { src = W2; K = HID;    base = dW1 + dU; }
  else if (d <  dW1 + 3 * dU) { src = U2; K = HID;    base = dW1 + 2 * dU; }
  else return;

  long idx    = d - base;
  int  KT     = K / 32;
  long tile   = idx >> 8;          // 256 dwords per tile
  int  within = (int)(idx & 255);
  int  lane   = within >> 3;
  int  reg    = within & 7;
  int  kt     = (int)(tile % KT);
  long nt     = tile / KT;
  int  n      = (int)(nt * 16) + (lane & 15);
  int  half   = lane >> 4;
  int  k0     = kt * 32 + (reg >> 2) * 16 + half * 8 + 2 * (reg & 3);

  float s0 = src[(long)k0 * GDIM + n];
  float s1 = src[(long)(k0 + 1) * GDIM + n];
  union { _Float16 h[2]; unsigned u; } pk;
  pk.h[0] = (_Float16)s0;
  pk.h[1] = (_Float16)s1;
  ((unsigned*)dst)[d] = pk.u;
}

// ---------------------------------------------------------------------------
// Fragment loaders
// ---------------------------------------------------------------------------
__device__ __forceinline__ v16h load_a_frag(const _Float16* base, int stride,
                                            int lane, int kt)
{
  // A 16x32 f16 layout: lane = M (per half), regs0-3: K = half*8.., regs4-7: K = 16+half*8..
  const int row = lane & 15, half = lane >> 4;
  const _Float16* p = base + row * stride + kt * 32;
  f16frag u;
  u.i[0] = *(const i32x4*)(p + half * 8);        // ds_load_b128
  u.i[1] = *(const i32x4*)(p + 16 + half * 8);   // ds_load_b128
  return u.v;
}

__device__ __forceinline__ v16h load_b_frag(const _Float16* tile, int lane)
{
  const i32x4* p = (const i32x4*)(tile + lane * 16);  // 32B per lane, coalesced
  f16frag u;
  u.i[0] = p[0];                                      // global_load_b128
  u.i[1] = p[1];                                      // global_load_b128
  return u.v;
}

__device__ __forceinline__ float sigf(float x) { return 1.0f / (1.0f + __expf(-x)); }

// ---------------------------------------------------------------------------
// One LSTM cell update for this block's batch tile.
// Each wave owns hidden slice [wave*64, wave*64+64), processed in 4 sub-slices
// of 16 columns; for each sub-slice it computes the i/f/o/cbar gate columns
// (n, n+512, n+1024, n+1536) so gate math is element-wise on its own regs.
// ---------------------------------------------------------------------------
template <int KT1>
__device__ void lstm_layer(
    const _Float16* __restrict__ A1, int a1_stride,   // first input (x or h1) in LDS
    const _Float16* __restrict__ Hprev,               // recurrent h in LDS (stride H_STR)
    const _Float16* __restrict__ Wp,                  // packed input weights (global/L2)
    const _Float16* __restrict__ Up,                  // packed recurrent weights
    const float* __restrict__ bB, const float* __restrict__ bU,
    float* __restrict__ cbase, _Float16* __restrict__ hout,
    int wave, int lane)
{
  const int half = lane >> 4;
  const int nlo  = lane & 15;

#pragma unroll 1
  for (int sub = 0; sub < 4; ++sub) {
    const int ntcol  = wave * 4 + sub;   // 16-col hidden tile index (0..31)
    const int colhid = ntcol * 16;

    v8f acc[2][4];
#pragma unroll
    for (int g = 0; g < 4; ++g) {
      const int col = g * HID + colhid + nlo;
      const float bv = bB[col] + bU[col];
#pragma unroll
      for (int v = 0; v < 8; ++v) { acc[0][g][v] = bv; acc[1][g][v] = bv; }
    }

    // --- GEMM 1: input contribution (K = KT1*32) ---
#pragma unroll 2
    for (int kt = 0; kt < KT1; ++kt) {
      v16h a0 = load_a_frag(A1, a1_stride, lane, kt);
      v16h a1 = load_a_frag(A1 + 16 * a1_stride, a1_stride, lane, kt);
#pragma unroll
      for (int g = 0; g < 4; ++g) {
        const _Float16* tile = Wp + ((long)(g * 32 + ntcol) * KT1 + kt) * 512;
        v16h b = load_b_frag(tile, lane);
        acc[0][g] = __builtin_amdgcn_wmma_f32_16x16x32_f16(false, a0, false, b,
                                                           (short)0, acc[0][g], false, false);
        acc[1][g] = __builtin_amdgcn_wmma_f32_16x16x32_f16(false, a1, false, b,
                                                           (short)0, acc[1][g], false, false);
      }
    }

    // --- GEMM 2: recurrent contribution (K = 512 -> 16 k-tiles) ---
#pragma unroll 2
    for (int kt = 0; kt < 16; ++kt) {
      v16h a0 = load_a_frag(Hprev, H_STR, lane, kt);
      v16h a1 = load_a_frag(Hprev + 16 * H_STR, H_STR, lane, kt);
#pragma unroll
      for (int g = 0; g < 4; ++g) {
        const _Float16* tile = Up + ((long)(g * 32 + ntcol) * 16 + kt) * 512;
        v16h b = load_b_frag(tile, lane);
        acc[0][g] = __builtin_amdgcn_wmma_f32_16x16x32_f16(false, a0, false, b,
                                                           (short)0, acc[0][g], false, false);
        acc[1][g] = __builtin_amdgcn_wmma_f32_16x16x32_f16(false, a1, false, b,
                                                           (short)0, acc[1][g], false, false);
      }
    }

    // --- gate application: element-wise over the 4 gate accumulators ---
    const int hcol = colhid + nlo;
#pragma unroll
    for (int rt = 0; rt < 2; ++rt) {
#pragma unroll
      for (int v = 0; v < 8; ++v) {
        const int row = rt * 16 + v + half * 8;   // C/D layout: M = v (+8 for hi half)
        const float gi = acc[rt][0][v];
        const float gf = acc[rt][1][v];
        const float go = acc[rt][2][v];
        const float gc = acc[rt][3][v];
        float* cp = cbase + row * C_STR + hcol;
        const float cold = *cp;
        const float cnew = sigf(gf) * cold + sigf(gi) * tanhf(gc);
        *cp = cnew;
        hout[row * H_STR + hcol] = (_Float16)(sigf(go) * tanhf(cnew));
      }
    }
  }
}

// ---------------------------------------------------------------------------
// Main persistent-per-batch-tile kernel: one block = 32 batch rows, full T loop.
// ---------------------------------------------------------------------------
__global__ void __launch_bounds__(256) lstm_main(
    const float* __restrict__ seq,
    const float* __restrict__ b1, const float* __restrict__ ub1,
    const float* __restrict__ b2, const float* __restrict__ ub2,
    const float* __restrict__ Wout, const float* __restrict__ bout,
    const _Float16* __restrict__ wpack,
    float* __restrict__ out)
{
  extern __shared__ char smem[];
  _Float16* xs   = (_Float16*)smem;
  _Float16* h1b  = (_Float16*)(smem + OFF_H);
  _Float16* h2b  = (_Float16*)(smem + OFF_H + 2 * HBUF_BYTES);
  float*    c1   = (float*)(smem + OFF_C);
  float*    c2   = (float*)(smem + OFF_C + CBUF_BYTES);
  float*    lred = (float*)(smem + OFF_LRED);

  const int tid  = threadIdx.x;
  const int wave = tid >> 5;
  const int lane = tid & 31;
  const int b0   = blockIdx.x * BT;

  const _Float16* W1p = wpack + W1P_OFF;
  const _Float16* U1p = wpack + U1P_OFF;
  const _Float16* W2p = wpack + W2P_OFF;
  const _Float16* U2p = wpack + U2P_OFF;

  // zero h (both buffers) and c
  {
    int* z = (int*)(smem + OFF_H);
    const int ndw = (4 * HBUF_BYTES + 2 * CBUF_BYTES) / 4;
    for (int i = tid; i < ndw; i += 256) z[i] = 0;
  }
  float pl0 = 0.0f, pl1 = 0.0f;
  __syncthreads();

  for (int t = 0; t < T_STEPS; ++t) {
    // ---- stage x_t tile into LDS as f16 (coalesced float4 reads) ----
    {
      const float* xsrc = seq + ((long)t * BATCH + b0) * IN_DIM;
#pragma unroll
      for (int it = 0; it < (BT * IN_DIM) / 1024; ++it) {
        const int idx = it * 1024 + tid * 4;
        const int row = idx >> 8, col = idx & 255;
        const float4 vv = *(const float4*)(xsrc + row * IN_DIM + col);
        union { _Float16 h[4]; unsigned long long u; } p4;
        p4.h[0] = (_Float16)vv.x; p4.h[1] = (_Float16)vv.y;
        p4.h[2] = (_Float16)vv.z; p4.h[3] = (_Float16)vv.w;
        *(unsigned long long*)(xs + row * XS_STR + col) = p4.u;
      }
    }
    __syncthreads();

    _Float16* h1cur  = h1b + (t & 1) * (BT * H_STR);
    _Float16* h1prev = h1b + ((t & 1) ^ 1) * (BT * H_STR);
    _Float16* h2cur  = h2b + (t & 1) * (BT * H_STR);
    _Float16* h2prev = h2b + ((t & 1) ^ 1) * (BT * H_STR);

    lstm_layer<IN_DIM / 32>(xs, XS_STR, h1prev, W1p, U1p, b1, ub1, c1, h1cur, wave, lane);
    __syncthreads();   // all of h1cur written
    lstm_layer<HID / 32>(h1cur, H_STR, h2prev, W2p, U2p, b2, ub2, c2, h2cur, wave, lane);
    __syncthreads();   // all of h2cur written

    // ---- readout: logits += h2[t] @ Wout[t*H : (t+1)*H, :] ----
    {
      const int row = tid >> 3, chunk = tid & 7;
      const float*    wr = Wout + ((long)t * HID + chunk * 64) * OUT_DIM;
      const _Float16* hr = h2cur + row * H_STR + chunk * 64;
#pragma unroll 4
      for (int j = 0; j < 64; ++j) {
        const float hv = (float)hr[j];
        pl0 += hv * wr[j * 2 + 0];
        pl1 += hv * wr[j * 2 + 1];
      }
    }
  }

  // ---- reduce partial logits, softmax, store ----
  lred[tid * 2 + 0] = pl0;
  lred[tid * 2 + 1] = pl1;
  __syncthreads();
  if ((tid & 7) == 0) {
    const int row = tid >> 3;
    float l0 = bout[0], l1 = bout[1];
#pragma unroll
    for (int k = 0; k < 8; ++k) {
      l0 += lred[(tid + k) * 2 + 0];
      l1 += lred[(tid + k) * 2 + 1];
    }
    const float m  = fmaxf(l0, l1);
    const float e0 = __expf(l0 - m), e1 = __expf(l1 - m);
    const float inv = 1.0f / (e0 + e1);
    out[(long)(b0 + row) * 2 + 0] = e0 * inv;
    out[(long)(b0 + row) * 2 + 1] = e1 * inv;
  }
}

// ---------------------------------------------------------------------------
// Host launcher
// ---------------------------------------------------------------------------
extern "C" void kernel_launch(void* const* d_in, const int* in_sizes, int n_in,
                              void* d_out, int out_size, void* d_ws, size_t ws_size,
                              hipStream_t stream)
{
  (void)in_sizes; (void)n_in; (void)out_size; (void)ws_size;

  const float* seq  = (const float*)d_in[0];
  const float* W1   = (const float*)d_in[1];
  const float* b1   = (const float*)d_in[2];
  const float* U1   = (const float*)d_in[3];
  const float* ub1  = (const float*)d_in[4];
  const float* W2   = (const float*)d_in[5];
  const float* b2   = (const float*)d_in[6];
  const float* U2   = (const float*)d_in[7];
  const float* ub2  = (const float*)d_in[8];
  const float* Wout = (const float*)d_in[9];
  const float* bout = (const float*)d_in[10];
  _Float16*    wpack = (_Float16*)d_ws;

  static bool attr_set = [] {
    hipFuncSetAttribute(reinterpret_cast<const void*>(lstm_main),
                        hipFuncAttributeMaxDynamicSharedMemorySize, SMEM_BYTES);
    return true;
  }();
  (void)attr_set;

  // 1) repack weights to f16 fragment tiles in scratch (cheap, deterministic)
  const long pack_dwords = PACK_F16_TOTAL / 2;      // 1,835,008
  prepack_kernel<<<(int)(pack_dwords / 256), 256, 0, stream>>>(W1, U1, W2, U2, wpack);

  // 2) full fused LSTM + readout + softmax
  lstm_main<<<BATCH / BT, 256, SMEM_BYTES, stream>>>(
      seq, b1, ub1, b2, ub2, Wout, bout, wpack, (float*)d_out);
}
